// ALiBiAttention_22050362097676
// MI455X (gfx1250) — matrix operand order repaired and checked by reading, not compiled, not run on hardware
//
#include <hip/hip_runtime.h>

typedef _Float16 v16h __attribute__((ext_vector_type(16)));
typedef _Float16 h8   __attribute__((ext_vector_type(8)));
typedef float    v8f  __attribute__((ext_vector_type(8)));

#define DMODEL 1024
#define NHEADS 16
#define HDIM   64
#define SEQ    2048
#define BATCH  2
#define MTOT   (BATCH*SEQ)   // 4096

__device__ __forceinline__ v8f wmma_f16(v16h a, v16h b, v8f c) {
    return __builtin_amdgcn_wmma_f32_16x16x32_f16(false, a, false, b, (short)0, c,
                                                  false, false);
}

// A fragment (16x32 f16, row-major, stride lda): lane half h holds K chunks
// [h*8 .. h*8+7] and [16+h*8 .. 16+h*8+7] of row M = lane&15.
__device__ __forceinline__ v16h load_afrag(const _Float16* base, int lda,
                                           int row0, int k0) {
    int lane = threadIdx.x & 31;
    int hf = lane >> 4;
    int m = lane & 15;
    const _Float16* p = base + (size_t)(row0 + m) * lda + k0 + hf * 8;
    union { v16h v; h8 h[2]; } u;
    u.h[0] = *(const h8*)(p);
    u.h[1] = *(const h8*)(p + 16);
    return u.v;
}

// B fragment (32x16) from B^T stored row-major [N][K] with stride ldbt:
// lane = N column, half h covers K = k0 + h*8 .. +7 and k0 + 16 + h*8 .. +7.
__device__ __forceinline__ v16h load_bfrag(const _Float16* baseT, int ldbt,
                                           int n0, int k0) {
    int lane = threadIdx.x & 31;
    int hf = lane >> 4;
    int n = lane & 15;
    const _Float16* p = baseT + (size_t)(n0 + n) * ldbt + k0 + hf * 8;
    union { v16h v; h8 h[2]; } u;
    u.h[0] = *(const h8*)(p);
    u.h[1] = *(const h8*)(p + 16);
    return u.v;
}

// ---------------- converts ----------------
__global__ void cvt_f16_kernel(const float* __restrict__ src,
                               _Float16* __restrict__ dst, int n) {
    int i = blockIdx.x * blockDim.x + threadIdx.x;
    if (i < n) dst[i] = (_Float16)src[i];
}

// dst[n*rows + k] = (f16) src[k*cols + n]   (W [K,N] -> W^T f16 [N,K])
__global__ void transpose_f16_kernel(const float* __restrict__ src,
                                     _Float16* __restrict__ dst,
                                     int rows, int cols) {
    int i = blockIdx.x * blockDim.x + threadIdx.x;
    if (i < rows * cols) {
        int n = i / rows;
        int k = i - n * rows;
        dst[i] = (_Float16)src[(size_t)k * cols + n];
    }
}

// ---------------- QKV projection GEMM ----------------
// wave: 32(M) x 64(N) tile (two A-row frags share each B frag -> 8 WMMA / k-step)
// block: 4 waves stacked in M (128 rows); blockIdx.z selects Q/K/V
__global__ void __launch_bounds__(128)
qkv_gemm_kernel(const _Float16* __restrict__ xh,
                const _Float16* __restrict__ WqT, const _Float16* __restrict__ WkT,
                const _Float16* __restrict__ WvT,
                const float* __restrict__ bq, const float* __restrict__ bk,
                const float* __restrict__ bv,
                _Float16* __restrict__ Qh, _Float16* __restrict__ Kh,
                _Float16* __restrict__ Vt) {
    int wave = threadIdx.x >> 5;
    int lane = threadIdx.x & 31;
    int row0 = blockIdx.x * 128 + wave * 32;
    int col0 = blockIdx.y * 64;
    int mode = blockIdx.z;
    const _Float16* WT = (mode == 0) ? WqT : (mode == 1) ? WkT : WvT;
    const float* bias = (mode == 0) ? bq : (mode == 1) ? bk : bv;

    v8f acc[2][4] = {};
    for (int k0 = 0; k0 < DMODEL; k0 += 32) {
        v16h a0 = load_afrag(xh, DMODEL, row0, k0);
        v16h a1 = load_afrag(xh, DMODEL, row0 + 16, k0);
#pragma unroll
        for (int nt = 0; nt < 4; ++nt) {
            v16h b = load_bfrag(WT, DMODEL, col0 + nt * 16, k0);
            acc[0][nt] = wmma_f16(a0, b, acc[0][nt]);
            acc[1][nt] = wmma_f16(a1, b, acc[1][nt]);
        }
    }
    int hf = lane >> 4, nl = lane & 15;
#pragma unroll
    for (int mt = 0; mt < 2; ++mt) {
#pragma unroll
        for (int nt = 0; nt < 4; ++nt) {
#pragma unroll
            for (int r = 0; r < 8; ++r) {
                int m = row0 + mt * 16 + r + 8 * hf;   // [0,4096)
                int n = col0 + nt * 16 + nl;           // [0,1024)
                float v = acc[mt][nt][r] + bias[n];
                int b = m >> 11, t = m & (SEQ - 1);
                int h = n >> 6, d = n & 63;
                _Float16 hv = (_Float16)v;
                if (mode == 2) {
                    // V transposed per head: [B,H,D,T]
                    Vt[(((size_t)(b * NHEADS + h)) * HDIM + d) * SEQ + t] = hv;
                } else {
                    _Float16* dst = (mode == 0) ? Qh : Kh;
                    dst[(((size_t)(b * NHEADS + h)) * SEQ + t) * HDIM + d] = hv;
                }
            }
        }
    }
}

// ---------------- flash attention with ALiBi ----------------
// one wave per 16-row q tile; kv streamed in blocks of 32 up to causal bound
__global__ void __launch_bounds__(128)
attn_kernel(const _Float16* __restrict__ Qh, const _Float16* __restrict__ Kh,
            const _Float16* __restrict__ Vt, _Float16* __restrict__ Oh) {
    __shared__ __attribute__((aligned(16))) _Float16 pl[4][16 * 32];
    int wave = threadIdx.x >> 5;
    int lane = threadIdx.x & 31;
    int hf = lane >> 4;
    int nl = lane & 15;
    int w = blockIdx.x * 4 + wave;
    int qt = w & 127;                 // T/16 = 128 q tiles
    int h = (w >> 7) & (NHEADS - 1);
    int b = w >> 11;
    int q0 = qt * 16;

    const _Float16* Qp = Qh + ((size_t)(b * NHEADS + h)) * SEQ * HDIM;
    const _Float16* Kp = Kh + ((size_t)(b * NHEADS + h)) * SEQ * HDIM;
    const _Float16* Vp = Vt + ((size_t)(b * NHEADS + h)) * HDIM * SEQ;

    v16h aq0 = load_afrag(Qp, HDIM, q0, 0);
    v16h aq1 = load_afrag(Qp, HDIM, q0, 32);

    v8f o[4] = {};
    float mrow[8], lrow[8];
#pragma unroll
    for (int r = 0; r < 8; ++r) { mrow[r] = -1e30f; lrow[r] = 0.0f; }

    const float scale = 0.125f;                        // 64^-0.5
    const float slope = exp2f(-0.5f * (float)(h + 1)); // ALiBi slope, H=16
    _Float16* pls = pl[wave];

    for (int kb = 0; kb < q0 + 16; kb += 32) {
        // prefetch next kv block (K rows: 128B each; V rows: d-major)
        if (kb + 32 < q0 + 16) {
            __builtin_prefetch(Kp + (size_t)(kb + 32 + lane) * HDIM, 0, 3);
            __builtin_prefetch(Vp + (size_t)(lane * 2) * SEQ + kb + 32, 0, 3);
            __builtin_prefetch(Vp + (size_t)(lane * 2 + 1) * SEQ + kb + 32, 0, 3);
        }
        // S tiles (16x16 each) for j ranges [kb,kb+16) and [kb+16,kb+32)
        v8f s[2];
#pragma unroll
        for (int jt = 0; jt < 2; ++jt) {
            v16h b0 = load_bfrag(Kp, HDIM, kb + jt * 16, 0);   // K^T tile, d 0..31
            v16h b1 = load_bfrag(Kp, HDIM, kb + jt * 16, 32);  // d 32..63
            v8f z = {};
            z = wmma_f16(aq0, b0, z);
            s[jt] = wmma_f16(aq1, b1, z);
        }
        // scale + ALiBi + causal mask (C-layout: reg r -> row r+8*hf, col nl)
#pragma unroll
        for (int jt = 0; jt < 2; ++jt) {
            int j = kb + jt * 16 + nl;
#pragma unroll
            for (int r = 0; r < 8; ++r) {
                int i = q0 + r + 8 * hf;
                float v = s[jt][r] * scale + slope * (float)(i - j);
                s[jt][r] = (j <= i) ? v : -3.0e38f;
            }
        }
        // online softmax: per-row max (16 lanes of each half share a row)
        float alpha[8];
#pragma unroll
        for (int r = 0; r < 8; ++r) {
            float mx = fmaxf(s[0][r], s[1][r]);
            mx = fmaxf(mx, __shfl_xor(mx, 1));
            mx = fmaxf(mx, __shfl_xor(mx, 2));
            mx = fmaxf(mx, __shfl_xor(mx, 4));
            mx = fmaxf(mx, __shfl_xor(mx, 8));
            float mnew = fmaxf(mrow[r], mx);
            alpha[r] = __expf(mrow[r] - mnew);
            mrow[r] = mnew;
        }
        // P = exp(S - m), row sums, stash f16 P tile in per-wave LDS
#pragma unroll
        for (int r = 0; r < 8; ++r) {
            float p0 = __expf(s[0][r] - mrow[r]);
            float p1 = __expf(s[1][r] - mrow[r]);
            float sum = p0 + p1;
            sum += __shfl_xor(sum, 1);
            sum += __shfl_xor(sum, 2);
            sum += __shfl_xor(sum, 4);
            sum += __shfl_xor(sum, 8);
            lrow[r] = lrow[r] * alpha[r] + sum;
            int row = r + 8 * hf;
            pls[row * 32 + nl] = (_Float16)p0;
            pls[row * 32 + 16 + nl] = (_Float16)p1;
        }
        // rescale O accumulators
#pragma unroll
        for (int dt = 0; dt < 4; ++dt)
#pragma unroll
            for (int r = 0; r < 8; ++r)
                o[dt][r] *= alpha[r];
        // reload P as an A-fragment (D-layout -> A-layout via LDS)
        v16h pf;
        {
            const _Float16* p = pls + nl * 32 + hf * 8;
            union { v16h v; h8 hh[2]; } u;
            u.hh[0] = *(const h8*)(p);
            u.hh[1] = *(const h8*)(p + 16);
            pf = u.v;
        }
        // O += P x V  (V^T layout makes B fragments contiguous)
#pragma unroll
        for (int dt = 0; dt < 4; ++dt) {
            v16h vb = load_bfrag(Vp, SEQ, dt * 16, kb);
            o[dt] = wmma_f16(pf, vb, o[dt]);
        }
    }
    // normalize + store f16 attention output [B*T, C]
#pragma unroll
    for (int dt = 0; dt < 4; ++dt) {
#pragma unroll
        for (int r = 0; r < 8; ++r) {
            int t = q0 + r + 8 * hf;
            int c = h * HDIM + dt * 16 + nl;
            float v = o[dt][r] / lrow[r];
            Oh[((size_t)(b * SEQ + t)) * DMODEL + c] = (_Float16)v;
        }
    }
}

// ---------------- output projection: out = Oh @ Wo + bo (f32) ----------------
// wave: 32(M) x 64(N) tile, same structure as qkv_gemm
__global__ void __launch_bounds__(128)
oproj_kernel(const _Float16* __restrict__ Oh, const _Float16* __restrict__ WoT,
             const float* __restrict__ bo, float* __restrict__ out) {
    int wave = threadIdx.x >> 5;
    int lane = threadIdx.x & 31;
    int row0 = blockIdx.x * 128 + wave * 32;
    int col0 = blockIdx.y * 64;

    v8f acc[2][4] = {};
    for (int k0 = 0; k0 < DMODEL; k0 += 32) {
        v16h a0 = load_afrag(Oh, DMODEL, row0, k0);
        v16h a1 = load_afrag(Oh, DMODEL, row0 + 16, k0);
#pragma unroll
        for (int nt = 0; nt < 4; ++nt) {
            v16h b = load_bfrag(WoT, DMODEL, col0 + nt * 16, k0);
            acc[0][nt] = wmma_f16(a0, b, acc[0][nt]);
            acc[1][nt] = wmma_f16(a1, b, acc[1][nt]);
        }
    }
    int hf = lane >> 4, nl = lane & 15;
#pragma unroll
    for (int mt = 0; mt < 2; ++mt) {
#pragma unroll
        for (int nt = 0; nt < 4; ++nt) {
#pragma unroll
            for (int r = 0; r < 8; ++r) {
                int m = row0 + mt * 16 + r + 8 * hf;
                int n = col0 + nt * 16 + nl;
                out[(size_t)m * DMODEL + n] = acc[mt][nt][r] + bo[n];
            }
        }
    }
}

extern "C" void kernel_launch(void* const* d_in, const int* in_sizes, int n_in,
                              void* d_out, int out_size, void* d_ws, size_t ws_size,
                              hipStream_t stream) {
    const float* x  = (const float*)d_in[0];
    const float* Wq = (const float*)d_in[1];
    const float* bq = (const float*)d_in[2];
    const float* Wk = (const float*)d_in[3];
    const float* bk = (const float*)d_in[4];
    const float* Wv = (const float*)d_in[5];
    const float* bv = (const float*)d_in[6];
    const float* Wo = (const float*)d_in[7];
    const float* bo = (const float*)d_in[8];

    char* ws = (char*)d_ws;
    _Float16* xh  = (_Float16*)(ws + ((size_t)0  << 20));  // 8 MB
    _Float16* WqT = (_Float16*)(ws + ((size_t)8  << 20));  // 2 MB
    _Float16* WkT = (_Float16*)(ws + ((size_t)10 << 20));  // 2 MB
    _Float16* WvT = (_Float16*)(ws + ((size_t)12 << 20));  // 2 MB
    _Float16* WoT = (_Float16*)(ws + ((size_t)14 << 20));  // 2 MB
    _Float16* Qh  = (_Float16*)(ws + ((size_t)16 << 20));  // 8 MB
    _Float16* Kh  = (_Float16*)(ws + ((size_t)24 << 20));  // 8 MB
    _Float16* Vt  = (_Float16*)(ws + ((size_t)32 << 20));  // 8 MB
    _Float16* Oh  = (_Float16*)(ws + ((size_t)40 << 20));  // 8 MB

    int nx = MTOT * DMODEL;          // 4,194,304
    cvt_f16_kernel<<<nx / 256, 256, 0, stream>>>(x, xh, nx);
    int nw = DMODEL * DMODEL;        // 1,048,576
    transpose_f16_kernel<<<nw / 256, 256, 0, stream>>>(Wq, WqT, DMODEL, DMODEL);
    transpose_f16_kernel<<<nw / 256, 256, 0, stream>>>(Wk, WkT, DMODEL, DMODEL);
    transpose_f16_kernel<<<nw / 256, 256, 0, stream>>>(Wv, WvT, DMODEL, DMODEL);
    transpose_f16_kernel<<<nw / 256, 256, 0, stream>>>(Wo, WoT, DMODEL, DMODEL);

    dim3 gq(MTOT / 128, DMODEL / 64, 3);   // 32 x 16 x 3
    qkv_gemm_kernel<<<gq, 128, 0, stream>>>(xh, WqT, WkT, WvT, bq, bk, bv,
                                            Qh, Kh, Vt);

    int nwaves = BATCH * NHEADS * (SEQ / 16);   // 4096
    attn_kernel<<<nwaves / 4, 128, 0, stream>>>(Qh, Kh, Vt, Oh);

    dim3 go(MTOT / 128, DMODEL / 64);      // 32 x 16
    oproj_kernel<<<go, 128, 0, stream>>>(Oh, WoT, bo, (float*)d_out);
}
